// SFM_WITRAN_2DPSGMU_Encoder_24034636988675
// MI455X (gfx1250) — matrix operand order repaired and checked by reading, not compiled, not run on hardware
//
#include <hip/hip_runtime.h>
#include <hip/hip_bf16.h>

typedef __attribute__((ext_vector_type(16))) _Float16 v16h;
typedef __attribute__((ext_vector_type(8)))  float    v8f;

#define NB    1024   // SN*BSZ = 32*32
#define LSTEP 63     // SN+OSL-1
#define HH    64
#define FQ_   32
#define IN_   16
#define KPAD  160    // 144 padded to 5 K-steps of 32
#define NCOL  448    // 384 gate cols + 64 freq cols
#define NT1   28     // 448/16 N-tiles
#define KS1   5      // 160/32 K-steps
#define OUT_MAIN  8257536   // 1024*63*128
#define OUT_HROW  8257536
#define OUT_HCOL  8323072   // + 65536

__device__ __forceinline__ float sigf(float x) { return 1.0f / (1.0f + __expf(-x)); }

// ---------------------------------------------------------------------------
// Pack fp32 weights into f16 WMMA B-fragment layout.
// B-frag (K=32 x N=16, f16): lane = n + 16*(k>=16), element e -> klocal = 16*(lane>>4)+e
// Linear index: ((nt*KS + s)*32 + lane)*16 + e
// ---------------------------------------------------------------------------
__global__ void witran_pack_weights(const float* __restrict__ Wf,   // 384 x 144
                                    const float* __restrict__ Wq,   // 64 x 144
                                    const float* __restrict__ roW,  // 64 x 64
                                    const float* __restrict__ coW,  // 64 x 64
                                    _Float16* __restrict__ Wcomb,   // 28*5*512
                                    _Float16* __restrict__ rowW,    // 4*2*512
                                    _Float16* __restrict__ colW) {  // 4*2*512
  const int stride = gridDim.x * blockDim.x;
  const int i0 = blockIdx.x * blockDim.x + threadIdx.x;
  for (int k = i0; k < NT1 * KS1 * 512; k += stride) {
    int e    = k & 15;
    int lane = (k >> 4) & 31;
    int rest = k >> 9;
    int s    = rest % KS1;
    int nt   = rest / KS1;
    int ncol = nt * 16 + (lane & 15);
    int kcol = s * 32 + ((lane >> 4) << 4) + e;
    float v = 0.0f;
    if (kcol < 144)
      v = (ncol < 384) ? Wf[ncol * 144 + kcol] : Wq[(ncol - 384) * 144 + kcol];
    Wcomb[k] = (_Float16)v;
  }
  for (int k = i0; k < 4 * 2 * 512; k += stride) {
    int e    = k & 15;
    int lane = (k >> 4) & 31;
    int s    = (k >> 9) & 1;
    int nt   = k >> 10;
    int ncol = nt * 16 + (lane & 15);
    int kcol = s * 32 + ((lane >> 4) << 4) + e;
    rowW[k] = (_Float16)roW[ncol * 64 + kcol];   // B[k=h][n=j] = row_o_w[j,h]
    colW[k] = (_Float16)coW[ncol * 64 + kcol];
  }
}

__global__ void witran_init_states(float* hrow0, float* hcol0,
                                   float* rr0, float* ri0, float* cr0, float* ci0) {
  const int stride = gridDim.x * blockDim.x;
  const int i0 = blockIdx.x * blockDim.x + threadIdx.x;
  for (int k = i0; k < NB * HH; k += stride) { hrow0[k] = 0.0f; hcol0[k] = 0.0f; }
  for (int k = i0; k < NB * HH * FQ_; k += stride) {
    rr0[k] = 0.1f; ri0[k] = 0.1f; cr0[k] = 0.1f; ci0[k] = 0.1f;
  }
}

// ---------------------------------------------------------------------------
// One scan step. 64 blocks x 128 threads; block handles 16 rows of NB.
// ---------------------------------------------------------------------------
__global__ __launch_bounds__(128)
void witran_step(const float* __restrict__ x,
                 const float* __restrict__ Bgate,   // 384
                 const float* __restrict__ Bfreq,   // 64
                 const float* __restrict__ rcw, const float* __restrict__ rcb,
                 const float* __restrict__ ccw, const float* __restrict__ ccb,
                 const float* __restrict__ rob, const float* __restrict__ cob,
                 const _Float16* __restrict__ Wcomb,
                 const _Float16* __restrict__ rowW,
                 const _Float16* __restrict__ colW,
                 const float* __restrict__ hrow_in, const float* __restrict__ hcol_in,
                 const float* __restrict__ rr_in,  const float* __restrict__ ri_in,
                 const float* __restrict__ cr_in,  const float* __restrict__ ci_in,
                 float* __restrict__ hrow_out, float* __restrict__ hcol_out,
                 float* __restrict__ rr_out,  float* __restrict__ ri_out,
                 float* __restrict__ cr_out,  float* __restrict__ ci_out,
                 float* __restrict__ out, int t) {
  __shared__ _Float16 sXcat[16][KPAD];   // f16 activations, K zero-padded
  __shared__ float    sGate[16][NCOL];   // raw GEMM output (pre-bias)
  __shared__ float    sFreq[16][64];     // sigmoided freq gates
  __shared__ float    sCos[16][FQ_];
  __shared__ float    sSin[16][FQ_];
  __shared__ _Float16 sRC[16][HH];       // row_c as f16 (A-matrix of GEMM2)
  __shared__ _Float16 sCC[16][HH];
  __shared__ float    sMM[16][128];      // [row_c@roW^T | col_c@coW^T]

  const int tid  = threadIdx.x;
  const int r0   = blockIdx.x * 16;
  const int wave = tid >> 5;
  const int lane = tid & 31;
  const int m    = lane & 15;
  const int hi   = lane >> 4;

  // ---- Phase 0: build xcat = [h_row | h_col | a_s] (f16) + trig tables ----
  for (int i = tid; i < 16 * KPAD; i += 128) {
    int row = i / KPAD, k = i % KPAD;
    int g = r0 + row;
    float v = 0.0f;
    if (k < 64)       v = hrow_in[g * 64 + k];
    else if (k < 128) v = hcol_in[g * 64 + (k - 64)];
    else if (k < 144) {
      int r = g >> 5, b = g & 31, c = t - r;
      if (c >= 0 && c < 32) v = x[(((b * 32) + r) * 32 + c) * 16 + (k - 128)];
    }
    sXcat[row][k] = (_Float16)v;
  }
  for (int i = tid; i < 16 * FQ_; i += 128) {
    int row = i >> 5, f = i & 31;
    int g = r0 + row;
    int rp = ((g >> 5) - t) & 31;           // (r - t) mod SN
    int cc2 = t - rp;
    float tm = (cc2 >= 0 && cc2 < 32) ? (float)(rp * 32 + cc2) : 0.0f;
    float om = 6.283185307f * (float)(f + 1) * (1.0f / 32.0f);
    sCos[row][f] = __cosf(tm * om);
    sSin[row][f] = __sinf(tm * om);
  }
  __syncthreads();

  // ---- Phase 1: gate GEMM  xcat(16x160) @ Wcomb^T -> sGate(16x448) ----
  v16h afr[KS1];
  #pragma unroll
  for (int s = 0; s < KS1; ++s) {
    v16h a;
    #pragma unroll
    for (int e = 0; e < 16; ++e) {
      int k = e + (hi << 4) + (s << 5);
      a[e] = sXcat[m][k];
    }
    afr[s] = a;
  }
  for (int j = 0; j < 7; ++j) {
    int nt = wave * 7 + j;
    v8f acc = {};
    #pragma unroll
    for (int s = 0; s < KS1; ++s) {
      const v16h b = *(const v16h*)(Wcomb + (((nt * KS1 + s) * 32 + lane) << 4));
      acc = __builtin_amdgcn_wmma_f32_16x16x32_f16(false, afr[s], false, b,
                                                   (short)0, acc, false, false);
    }
    int ncol = nt * 16 + (lane & 15);
    #pragma unroll
    for (int v = 0; v < 8; ++v) sGate[hi * 8 + v][ncol] = acc[v];
  }
  __syncthreads();

  // ---- Phase 2: sigmoided frequency gates (bias masked by bmask) ----
  for (int i = tid; i < 16 * 64; i += 128) {
    int row = i >> 6, fc = i & 63;
    int g = r0 + row;
    float bmv = (t < 32 && g < 32 * (t + 1)) ? 1.0f : 0.0f;
    sFreq[row][fc] = sigf(sGate[row][384 + fc] + bmv * Bfreq[fc]);
  }
  __syncthreads();

  // ---- Phase 3: spectral state update + Fq contraction (L2-resident) ----
  #pragma unroll
  for (int p = 0; p < 8; ++p) {
    int idx = tid + p * 128;
    int row = idx >> 6, h = idx & 63;
    int g = r0 + row;
    float bmv = (t < 32 && g < 32 * (t + 1)) ? 1.0f : 0.0f;
    float ugr = sigf (sGate[row][h]        + bmv * Bgate[h]);
    float ogr = sigf (sGate[row][64 + h]   + bmv * Bgate[64 + h]);
    float ugc = sigf (sGate[row][128 + h]  + bmv * Bgate[128 + h]);
    float ogc = sigf (sGate[row][192 + h]  + bmv * Bgate[192 + h]);
    float igr = tanhf(sGate[row][256 + h]  + bmv * Bgate[256 + h]);
    float igc = tanhf(sGate[row][320 + h]  + bmv * Bgate[320 + h]);
    float ampr = ogr * igr, ampc = ogc * igc;
    size_t base  = ((size_t)g * 64 + h) * 32;
    size_t baseR = ((size_t)((g + 32) & (NB - 1)) * 64 + h) * 32;   // rolled
    float sr = 0.0f, sc = 0.0f;
    #pragma unroll 4
    for (int f = 0; f < 32; ++f) {
      float cv = sCos[row][f], sv = sSin[row][f];
      float rftv = ugr * sFreq[row][f];
      float cftv = ugc * sFreq[row][32 + f];
      float rrn = rftv * rr_in[base + f] + ampr * cv;
      float rin = rftv * ri_in[base + f] + ampr * sv;
      float crn = cftv * cr_in[base + f] + ampc * cv;
      float cin = cftv * ci_in[base + f] + ampc * sv;
      rr_out[base + f]  = rrn;
      ri_out[base + f]  = rin;
      cr_out[baseR + f] = crn;
      ci_out[baseR + f] = cin;
      sr += (rrn * rrn + rin * rin) * rcw[f];
      sc += (crn * crn + cin * cin) * ccw[f];
    }
    sRC[row][h] = (_Float16)tanhf(sr + rcb[0]);
    sCC[row][h] = (_Float16)tanhf(sc + ccb[0]);
  }
  __syncthreads();

  // ---- Phase 4: small GEMMs  row_c@roW^T, col_c@coW^T (WMMA f16) ----
  {
    const _Float16* Wp  = (wave < 2) ? rowW : colW;
    const _Float16* src = (wave < 2) ? &sRC[0][0] : &sCC[0][0];
    const int colbase   = (wave < 2) ? 0 : 64;
    const int w2        = wave & 1;
    v16h a2[2];
    #pragma unroll
    for (int s = 0; s < 2; ++s) {
      v16h a;
      #pragma unroll
      for (int e = 0; e < 16; ++e) {
        int k = e + (hi << 4) + (s << 5);
        a[e] = src[m * 64 + k];
      }
      a2[s] = a;
    }
    #pragma unroll
    for (int j = 0; j < 2; ++j) {
      int nt = w2 * 2 + j;
      v8f acc = {};
      #pragma unroll
      for (int s = 0; s < 2; ++s) {
        const v16h b = *(const v16h*)(Wp + (((nt * 2 + s) * 32 + lane) << 4));
        acc = __builtin_amdgcn_wmma_f32_16x16x32_f16(false, a2[s], false, b,
                                                     (short)0, acc, false, false);
      }
      int ncol = colbase + nt * 16 + (lane & 15);
      #pragma unroll
      for (int v = 0; v < 8; ++v) sMM[hi * 8 + v][ncol] = acc[v];
    }
  }
  __syncthreads();

  // ---- Phase 5: h updates + outputs ----
  #pragma unroll
  for (int p = 0; p < 8; ++p) {
    int idx = tid + p * 128;
    int row = idx >> 6, h = idx & 63;
    int g = r0 + row;
    float bmv = (t < 32 && g < 32 * (t + 1)) ? 1.0f : 0.0f;
    float ugr = sigf (sGate[row][h]       + bmv * Bgate[h]);
    float ugc = sigf (sGate[row][128 + h] + bmv * Bgate[128 + h]);
    float igr = tanhf(sGate[row][256 + h] + bmv * Bgate[256 + h]);
    float igc = tanhf(sGate[row][320 + h] + bmv * Bgate[320 + h]);
    float rc = (float)sRC[row][h];
    float cc = (float)sCC[row][h];
    float hr_old = hrow_in[g * 64 + h];
    float hc_old = hcol_in[g * 64 + h];
    float hr = sigf((1.0f - ugr) * hr_old + ugr * igr + sMM[row][h]      + rob[h]) * rc;
    float hc = sigf((1.0f - ugc) * hc_old + ugc * igc + sMM[row][64 + h] + cob[h]) * cc;
    hrow_out[g * 64 + h] = hr;
    hcol_out[((g + 32) & (NB - 1)) * 64 + h] = hc;                 // roll
    size_t ob = (size_t)g * (LSTEP * 128) + (size_t)t * 128;
    out[ob + h]      = hr;
    out[ob + 64 + h] = hc;
    if (t >= 31) {
      int j = t - 31;
      if ((g >> 5) == j) out[OUT_HROW + (((g & 31) * 32 + j) * 64) + h] = hr;
      if (g >= NB - 32)  out[OUT_HCOL + (((g - (NB - 32)) * 32 + j) * 64) + h] = hc;
    }
  }
}

extern "C" void kernel_launch(void* const* d_in, const int* in_sizes, int n_in,
                              void* d_out, int out_size, void* d_ws, size_t ws_size,
                              hipStream_t stream) {
  const float* x   = (const float*)d_in[0];
  const float* Wf  = (const float*)d_in[1];
  const float* Wq  = (const float*)d_in[2];
  const float* Bg  = (const float*)d_in[3];
  const float* Bq  = (const float*)d_in[4];
  const float* rcw = (const float*)d_in[5];
  const float* rcb = (const float*)d_in[6];
  const float* ccw = (const float*)d_in[7];
  const float* ccb = (const float*)d_in[8];
  const float* roW = (const float*)d_in[9];
  const float* rob = (const float*)d_in[10];
  const float* coW = (const float*)d_in[11];
  const float* cob = (const float*)d_in[12];

  char* ws = (char*)d_ws;
  _Float16* Wcomb = (_Float16*)(ws);                 // 143360 B
  _Float16* rowW  = (_Float16*)(ws + 143360);        // 8192 B
  _Float16* colW  = (_Float16*)(ws + 151552);        // 8192 B
  size_t off = 159744;
  float* hrow[2]; float* hcol[2]; float* rr[2]; float* ri[2]; float* cr[2]; float* ci[2];
  const size_t hbytes = (size_t)NB * HH * sizeof(float);            // 256 KB
  const size_t sbytes = (size_t)NB * HH * FQ_ * sizeof(float);      // 8 MB
  hrow[0] = (float*)(ws + off); off += hbytes;
  hrow[1] = (float*)(ws + off); off += hbytes;
  hcol[0] = (float*)(ws + off); off += hbytes;
  hcol[1] = (float*)(ws + off); off += hbytes;
  rr[0] = (float*)(ws + off); off += sbytes;
  rr[1] = (float*)(ws + off); off += sbytes;
  ri[0] = (float*)(ws + off); off += sbytes;
  ri[1] = (float*)(ws + off); off += sbytes;
  cr[0] = (float*)(ws + off); off += sbytes;
  cr[1] = (float*)(ws + off); off += sbytes;
  ci[0] = (float*)(ws + off); off += sbytes;
  ci[1] = (float*)(ws + off); off += sbytes;

  witran_pack_weights<<<64, 256, 0, stream>>>(Wf, Wq, roW, coW, Wcomb, rowW, colW);
  witran_init_states<<<256, 256, 0, stream>>>(hrow[0], hcol[0], rr[0], ri[0], cr[0], ci[0]);

  for (int t = 0; t < LSTEP; ++t) {
    int pi = t & 1, po = 1 - pi;
    witran_step<<<64, 128, 0, stream>>>(
        x, Bg, Bq, rcw, rcb, ccw, ccb, rob, cob,
        Wcomb, rowW, colW,
        hrow[pi], hcol[pi], rr[pi], ri[pi], cr[pi], ci[pi],
        hrow[po], hcol[po], rr[po], ri[po], cr[po], ci[po],
        (float*)d_out, t);
  }
  (void)in_sizes; (void)n_in; (void)out_size; (void)ws_size;
}
